// DCNv2_48996986913012
// MI455X (gfx1250) — compile-verified, matrix operand order checked by reading, not run on hardware
//
#include <hip/hip_runtime.h>
#include <math.h>

// ---------------------------------------------------------------------------
// MI455X (gfx1250) implementation of the DCNv2 CNN pipeline.
// All convolutions (plain + modulated-deformable) are implicit GEMMs using
// v_wmma_f32_16x16x32_bf16 (wave32). Activations live in NHWC fp32; each
// training-mode BatchNorm is folded into the consumer's im2col as a per-
// channel affine (scale/shift). Weights are pre-packed fragment-major so the
// B operand is two global_load_b128 per WMMA; the A operand comes from the
// LDS im2col tile as two ds_load_b128 per WMMA.
// ---------------------------------------------------------------------------

typedef __attribute__((ext_vector_type(16))) __bf16 v16bf;
typedef __attribute__((ext_vector_type(8)))  float  v8f;

#define NPIX   262144      // B*H*W = 64*64*64
#define HW     4096
#define BATCH  64

static __device__ __forceinline__ __bf16 f2bf(float f) { return (__bf16)f; }

// K-half index map for 16-bit WMMA fragments (ISA 7.12.2):
// lanes 0-15 hold K = 0..7 and 16..23 ; lanes 16-31 hold K = 8..15 and 24..31
static __host__ __device__ __forceinline__ int kmap(int hi, int h) {
    return (h < 8) ? (hi * 8 + h) : (16 + hi * 8 + (h - 8));
}

// ---------------------------------------------------------------------------
// Per-wave WMMA GEMM over a 64-pixel LDS im2col tile.
// sA  : [64, KP] bf16 (LDS, row-major; per-lane fragment = 2x contiguous 16B)
// wpk : fragment-major bf16: lane l of tile (kk,ni) at [((kk*NT+ni)*32+l)*16]
// ---------------------------------------------------------------------------
template<int KP, int NP, int COUT, bool RELU>
static __device__ __forceinline__ void gemm_tiles(const __bf16* sA,
                                                  const __bf16* __restrict__ wpk,
                                                  const float* __restrict__ bias,
                                                  float* __restrict__ out,
                                                  int p0, int tid)
{
    const int wave = tid >> 5;
    const int lane = tid & 31;
    const int lm   = lane & 15;
    const int hi   = lane >> 4;
    constexpr int NT = NP / 16;
    constexpr int KS = KP / 32;

    __builtin_prefetch(wpk, 0, 1);   // global_prefetch_b8: warm weights into L2

    for (int t = wave; t < 4 * NT; t += 8) {
        const int mi = t / NT;
        const int ni = t - mi * NT;
        v8f acc = {0.f, 0.f, 0.f, 0.f, 0.f, 0.f, 0.f, 0.f};
#pragma unroll 3
        for (int kk = 0; kk < KS; ++kk) {
            // A fragment: two contiguous 8-half runs per lane -> ds_load_b128 x2
            v16bf af;
#pragma unroll
            for (int h = 0; h < 16; ++h)
                af[h] = sA[(mi * 16 + lm) * KP + kk * 32 + kmap(hi, h)];
            // B fragment: fragment-major packed -> global_load_b128 x2
            const v16bf bf = *reinterpret_cast<const v16bf*>(
                wpk + (((size_t)(kk * NT + ni) * 32 + lane) << 4));
            acc = __builtin_amdgcn_wmma_f32_16x16x32_bf16(
                false, af, false, bf, (short)0, acc, false, false);
        }
        const int n = ni * 16 + lm;
        if (n < COUT) {
            const float bv = bias ? bias[n] : 0.0f;
#pragma unroll
            for (int r = 0; r < 8; ++r) {
                const int m = hi * 8 + r;                      // C/D: M = hi*8+r
                float v = acc[r] + bv;
                if (RELU) v = fmaxf(v, 0.0f);
                out[(size_t)(p0 + mi * 16 + m) * COUT + n] = v;
            }
        }
    }
}

// ---------------------------------------------------------------------------
// Generic 3x3 / stride 1 / pad 1 conv as implicit GEMM. Input NHWC fp32 with
// optional fused BN (bnp = [mean|rstd|gamma|beta] of size 4*CIN).
// ---------------------------------------------------------------------------
template<int CIN, int COUT, bool RELU, bool HASBN>
__global__ void conv_gemm(const float* __restrict__ in,
                          const float* __restrict__ bnp,
                          const __bf16* __restrict__ wpk,
                          const float* __restrict__ bias,
                          float* __restrict__ out)
{
    constexpr int KP = ((CIN * 9 + 31) / 32) * 32;
    constexpr int NP = ((COUT + 15) / 16) * 16;
    constexpr int MT = 64;
    __shared__ __bf16 sA[MT * KP];
    __shared__ float  sSc[CIN], sSh[CIN];

    const int tid = threadIdx.x;
    if (tid < CIN) {
        if (HASBN) {
            const float mn = bnp[tid], rs = bnp[CIN + tid];
            const float g = bnp[2 * CIN + tid], b = bnp[3 * CIN + tid];
            const float sc = g * rs;
            sSc[tid] = sc; sSh[tid] = b - mn * sc;
        } else { sSc[tid] = 1.f; sSh[tid] = 0.f; }
    }
    __syncthreads();

    const int p0 = blockIdx.x * MT;
    for (int e = tid; e < MT * KP; e += blockDim.x) {
        const int m = e / KP;
        const int k = e - m * KP;
        float v = 0.f;
        if (k < 9 * CIN) {
            const int tap = k / CIN;
            const int c   = k - tap * CIN;
            const int p = p0 + m;
            const int b = p >> 12, s = p & 4095;
            const int y = s >> 6,  x = s & 63;
            const int yy = y + tap / 3 - 1;
            const int xx = x + (tap - (tap / 3) * 3) - 1;
            if (yy >= 0 && yy < 64 && xx >= 0 && xx < 64) {
                const float raw = in[((size_t)(b * HW + yy * 64 + xx)) * CIN + c];
                v = raw * sSc[c] + sSh[c];
            }
        }
        sA[e] = f2bf(v);
    }
    __syncthreads();
    gemm_tiles<KP, NP, COUT, RELU>(sA, wpk, bias, out, p0, tid);
}

// ---------------------------------------------------------------------------
// Modulated deformable conv (DCNv2): bilinear gather of BN'd input modulated
// by sigmoid mask into LDS im2col, then the same WMMA GEMM (K = 9*CIN).
// om : NHWC [NPIX, 3*DG*9] = [offy | offx | mask-logits].
// ---------------------------------------------------------------------------
template<int CIN, int DG, int COUT>
__global__ void dconv_gemm(const float* __restrict__ in,
                           const float* __restrict__ bnp,
                           const float* __restrict__ om,
                           const __bf16* __restrict__ wpk,
                           const float* __restrict__ bias,
                           float* __restrict__ out)
{
    constexpr int Cg  = CIN / DG;
    constexpr int KD  = 9 * CIN;
    constexpr int KP  = ((KD + 31) / 32) * 32;
    constexpr int NP  = ((COUT + 15) / 16) * 16;
    constexpr int MT  = 64;
    constexpr int OMC = 3 * DG * 9;
    __shared__ __bf16 sA[MT * KP];
    __shared__ float  sSc[CIN], sSh[CIN];

    const int tid = threadIdx.x;
    if (tid < CIN) {
        const float mn = bnp[tid], rs = bnp[CIN + tid];
        const float g = bnp[2 * CIN + tid], b = bnp[3 * CIN + tid];
        const float sc = g * rs;
        sSc[tid] = sc; sSh[tid] = b - mn * sc;
    }
    __syncthreads();

    const int p0 = blockIdx.x * MT;
    // zero K-padding columns
    for (int e = tid; e < MT * (KP - KD); e += blockDim.x) {
        const int m = e / (KP - KD);
        const int k = KD + (e - m * (KP - KD));
        sA[m * KP + k] = f2bf(0.f);
    }
    // one job per (pixel, group, tap)
    for (int j = tid; j < MT * DG * 9; j += blockDim.x) {
        const int m   = j / (DG * 9);
        const int r   = j - m * (DG * 9);
        const int g   = r / 9;
        const int tap = r - g * 9;
        const int p = p0 + m;
        const int b = p >> 12, s = p & 4095;
        const int y = s >> 6,  x = s & 63;
        const float* omp = om + (size_t)p * OMC;
        const float offy = omp[g * 9 + tap];
        const float offx = omp[DG * 9 + g * 9 + tap];
        const float msk  = 1.f / (1.f + __expf(-omp[2 * DG * 9 + g * 9 + tap]));
        const float py = (float)(y + tap / 3 - 1) + offy;
        const float px = (float)(x + (tap % 3) - 1) + offx;
        const float y0f = floorf(py), x0f = floorf(px);
        const int y0 = (int)y0f, x0 = (int)x0f;
        const int y1 = y0 + 1,   x1 = x0 + 1;
        const float ly = py - y0f, lx = px - x0f;
        float w00 = (1.f - ly) * (1.f - lx);
        float w01 = (1.f - ly) * lx;
        float w10 = ly * (1.f - lx);
        float w11 = ly * lx;
        const bool vy0 = (y0 >= 0 && y0 < 64), vy1 = (y1 >= 0 && y1 < 64);
        const bool vx0 = (x0 >= 0 && x0 < 64), vx1 = (x1 >= 0 && x1 < 64);
        if (!(vy0 && vx0)) w00 = 0.f;
        if (!(vy0 && vx1)) w01 = 0.f;
        if (!(vy1 && vx0)) w10 = 0.f;
        if (!(vy1 && vx1)) w11 = 0.f;
        const int cy0 = vy0 ? y0 : 0, cy1 = vy1 ? y1 : 0;
        const int cx0 = vx0 ? x0 : 0, cx1 = vx1 ? x1 : 0;
        const size_t bb = (size_t)b * HW;
        const float* p00 = in + (bb + cy0 * 64 + cx0) * CIN + g * Cg;
        const float* p01 = in + (bb + cy0 * 64 + cx1) * CIN + g * Cg;
        const float* p10 = in + (bb + cy1 * 64 + cx0) * CIN + g * Cg;
        const float* p11 = in + (bb + cy1 * 64 + cx1) * CIN + g * Cg;
        __bf16* dst = &sA[m * KP + (g * 9 + tap) * Cg];
        for (int c = 0; c < Cg; ++c) {
            const int ch = g * Cg + c;
            const float sc = sSc[ch], sh = sSh[ch];
            const float v = w00 * (p00[c] * sc + sh) + w01 * (p01[c] * sc + sh)
                          + w10 * (p10[c] * sc + sh) + w11 * (p11[c] * sc + sh);
            dst[c] = f2bf(msk * v);
        }
    }
    __syncthreads();
    gemm_tiles<KP, NP, COUT, true>(sA, wpk, bias, out, p0, tid);
}

// ---------------------------------------------------------------------------
// Weight pre-pack: fp32 OIHW -> bf16 fragment-major B layout.
// Element i: h = i&15, lane = (i>>4)&31, tile = i>>9 (tile = kk*NT + ni).
// Lane's halves are contiguous (32B) -> B frag is one aligned v16bf load.
// ---------------------------------------------------------------------------
__global__ void pack_w(const float* __restrict__ w, __bf16* __restrict__ wpk,
                       int CIN, int COUT, int KP, int NP)
{
    const int i = blockIdx.x * blockDim.x + threadIdx.x;
    if (i >= KP * NP) return;
    const int h    = i & 15;
    const int lane = (i >> 4) & 31;
    const int tile = i >> 9;
    const int NT = NP / 16;
    const int ni = tile % NT;
    const int kk = tile / NT;
    const int n  = ni * 16 + (lane & 15);
    const int k  = kk * 32 + kmap(lane >> 4, h);
    float v = 0.f;
    if (k < 9 * CIN && n < COUT) {
        const int tap = k / CIN, c = k - tap * CIN;          // k = tap*CIN + c
        v = w[((size_t)n * CIN + c) * 9 + tap];
    }
    wpk[i] = f2bf(v);
}

__global__ void pack_dw(const float* __restrict__ w, __bf16* __restrict__ wpk,
                        int CIN, int DG, int COUT, int KP, int NP)
{
    const int i = blockIdx.x * blockDim.x + threadIdx.x;
    if (i >= KP * NP) return;
    const int h    = i & 15;
    const int lane = (i >> 4) & 31;
    const int tile = i >> 9;
    const int NT = NP / 16;
    const int ni = tile % NT;
    const int kk = tile / NT;
    const int n  = ni * 16 + (lane & 15);
    const int k  = kk * 32 + kmap(lane >> 4, h);
    const int Cg = CIN / DG;
    float v = 0.f;
    if (k < 9 * CIN && n < COUT) {
        const int gt = k / Cg, c = k - gt * Cg;              // k = (g*9+tap)*Cg + c
        const int g = gt / 9, tap = gt - g * 9;
        v = w[((size_t)n * CIN + (g * Cg + c)) * 9 + tap];
    }
    wpk[i] = f2bf(v);
}

// ---------------------------------------------------------------------------
// BatchNorm statistics: each lane owns channel(s) (c = lane, lane+32) and
// accumulates privately over pixels (coalesced NHWC rows), then one atomic
// per owned channel per wave.
// ---------------------------------------------------------------------------
template<int C>
__global__ void stats_accum(const float* __restrict__ a, float* __restrict__ sums)
{
    const int tid  = threadIdx.x;
    const int lane = tid & 31;
    const int gw = blockIdx.x * (blockDim.x >> 5) + (tid >> 5);
    const int nw = gridDim.x * (blockDim.x >> 5);
    float s0 = 0.f, q0 = 0.f, s1 = 0.f, q1 = 0.f;
    const int c0 = lane, c1 = lane + 32;
    for (int p = gw; p < NPIX; p += nw) {
        const float* row = a + (size_t)p * C;
        if (c0 < C) { const float v = row[c0]; s0 += v; q0 += v * v; }
        if (C > 32 && c1 < C) { const float v = row[c1]; s1 += v; q1 += v * v; }
    }
    if (c0 < C) { atomicAdd(&sums[c0], s0); atomicAdd(&sums[C + c0], q0); }
    if (C > 32 && c1 < C) { atomicAdd(&sums[c1], s1); atomicAdd(&sums[C + c1], q1); }
}

template<int C>
__global__ void stats_final(const float* __restrict__ sums,
                            const float* __restrict__ gam,
                            const float* __restrict__ bet,
                            float* __restrict__ bnp)
{
    const int c = threadIdx.x;
    if (c < C) {
        const float inv = 1.0f / (float)NPIX;
        const float m = sums[c] * inv;
        const float var = sums[C + c] * inv - m * m;
        bnp[c] = m;
        bnp[C + c] = rsqrtf(var + 1e-5f);
        bnp[2 * C + c] = gam[c];
        bnp[3 * C + c] = bet[c];
    }
}

// ---------------------------------------------------------------------------
// Layout shuffles + epilogue
// ---------------------------------------------------------------------------
__global__ void nchw_to_nhwc3(const float* __restrict__ x, float* __restrict__ y)
{
    const int e = blockIdx.x * blockDim.x + threadIdx.x;     // NPIX*3 elems
    if (e >= NPIX * 3) return;
    const int p = e / 3, c = e - p * 3;
    const int b = p >> 12, s = p & 4095;
    y[e] = x[((size_t)b * 3 + c) * HW + s];
}

__global__ void emit_em(const float* __restrict__ a4, const float* __restrict__ bnp,
                        float* __restrict__ em)
{
    const int e = blockIdx.x * blockDim.x + threadIdx.x;     // NPIX*64 elems
    const int p = e >> 6, c = e & 63;
    const int b = p >> 12, s = p & 4095;
    const float sc = bnp[2 * 64 + c] * bnp[64 + c];
    const float sh = bnp[3 * 64 + c] - bnp[c] * sc;
    em[((size_t)b * 64 + c) * HW + s] = a4[e] * sc + sh;
}

__global__ void pool_fc(const float* __restrict__ em, const float* __restrict__ fc_w,
                        const float* __restrict__ fc_b, float* __restrict__ out)
{
    const int b = blockIdx.x;       // one block per batch element
    const int t = threadIdx.x;      // 256 threads = 64ch * 2x2 quadrants
    __shared__ float pool[256];
    __shared__ float lg[10];
    const int c = t >> 2, qy = (t >> 1) & 1, qx = t & 1;
    const float* base = em + ((size_t)b * 64 + c) * HW + qy * 32 * 64 + qx * 32;
    float s = 0.f;
    for (int yy = 0; yy < 32; ++yy)
        for (int xx = 0; xx < 32; ++xx) s += base[yy * 64 + xx];
    pool[t] = s * (1.0f / 1024.0f);                          // pool idx = c*4+qy*2+qx
    __syncthreads();
    if (t < 10) {
        float acc = fc_b[t];
        for (int i = 0; i < 256; ++i) acc += pool[i] * fc_w[t * 256 + i];
        lg[t] = acc;
    }
    __syncthreads();
    if (t == 0) {
        float mx = lg[0];
        for (int j = 1; j < 10; ++j) mx = fmaxf(mx, lg[j]);
        float e[10], sum = 0.f;
        for (int j = 0; j < 10; ++j) { e[j] = expf(lg[j] - mx); sum += e[j]; }
        const float inv = 1.0f / sum;
        for (int j = 0; j < 10; ++j) out[b * 10 + j] = e[j] * inv;
    }
}

// ---------------------------------------------------------------------------
// Host launcher. Workspace layout (~166 MB peak, aliased regions):
//   A [64MB]: a2 (20ch) then a4 (64ch)      B [54MB]: om1 then om2
//   C [40MB]: xh(3ch)+a1(8ch) then a3(40ch) W: packed weights + stats + bnp
// ---------------------------------------------------------------------------
static inline size_t alup(size_t x) { return (x + 255) & ~(size_t)255; }

extern "C" void kernel_launch(void* const* d_in, const int* in_sizes, int n_in,
                              void* d_out, int out_size, void* d_ws, size_t ws_size,
                              hipStream_t stream)
{
    (void)in_sizes; (void)n_in; (void)out_size; (void)ws_size;
    const float* x       = (const float*)d_in[0];
    const float* conv1_w = (const float*)d_in[1];
    const float* bn1_g   = (const float*)d_in[2];
    const float* bn1_b   = (const float*)d_in[3];
    const float* conv2_w = (const float*)d_in[4];
    const float* conv2_b = (const float*)d_in[5];
    const float* bn2_g   = (const float*)d_in[6];
    const float* bn2_b   = (const float*)d_in[7];
    const float* off1_w  = (const float*)d_in[8];
    const float* off1_b  = (const float*)d_in[9];
    const float* d1_w    = (const float*)d_in[10];
    const float* d1_b    = (const float*)d_in[11];
    const float* bn3_g   = (const float*)d_in[12];
    const float* bn3_b   = (const float*)d_in[13];
    const float* off2_w  = (const float*)d_in[14];
    const float* off2_b  = (const float*)d_in[15];
    const float* d2_w    = (const float*)d_in[16];
    const float* d2_b    = (const float*)d_in[17];
    const float* bn4_g   = (const float*)d_in[18];
    const float* bn4_b   = (const float*)d_in[19];
    const float* fc_w    = (const float*)d_in[20];
    const float* fc_b    = (const float*)d_in[21];

    char* ws = (char*)d_ws;
    const size_t ofsA  = 0;                                  // 64-ch region
    const size_t ofsB  = (size_t)NPIX * 64 * 4;              // 54-ch region
    const size_t ofsC  = ofsB + (size_t)NPIX * 54 * 4;       // 40-ch region
    const size_t ofsW0 = ofsC + (size_t)NPIX * 40 * 4;

    float* a2  = (float*)(ws + ofsA);
    float* a4  = (float*)(ws + ofsA);
    float* om1 = (float*)(ws + ofsB);
    float* om2 = (float*)(ws + ofsB);
    float* xh  = (float*)(ws + ofsC);
    float* a1  = (float*)(ws + ofsC + (size_t)NPIX * 3 * 4);
    float* a3  = (float*)(ws + ofsC);

    size_t o = ofsW0;
    __bf16* w1pk  = (__bf16*)(ws + o); o = alup(o + 32  * 16 * 2);
    __bf16* w2pk  = (__bf16*)(ws + o); o = alup(o + 96  * 32 * 2);
    __bf16* wo1pk = (__bf16*)(ws + o); o = alup(o + 192 * 64 * 2);
    __bf16* wd1pk = (__bf16*)(ws + o); o = alup(o + 192 * 48 * 2);
    __bf16* wo2pk = (__bf16*)(ws + o); o = alup(o + 384 * 64 * 2);
    __bf16* wd2pk = (__bf16*)(ws + o); o = alup(o + 384 * 64 * 2);
    float*  sums  = (float*)(ws + o);  o = alup(o + 2 * 64 * 4);
    float*  bnp1  = (float*)(ws + o);  o = alup(o + 4 * 8  * 4);
    float*  bnp2  = (float*)(ws + o);  o = alup(o + 4 * 20 * 4);
    float*  bnp3  = (float*)(ws + o);  o = alup(o + 4 * 40 * 4);
    float*  bnp4  = (float*)(ws + o);  o = alup(o + 4 * 64 * 4);

    float* out_sm = (float*)d_out;            // [64,10]
    float* out_em = (float*)d_out + 640;      // [64,64,64,64] NCHW

    // ---- weight packing (fragment-major) --------------------------------
    pack_w <<<(32 * 16  + 255) / 256, 256, 0, stream>>>(conv1_w, w1pk,  3,  8,  32, 16);
    pack_w <<<(96 * 32  + 255) / 256, 256, 0, stream>>>(conv2_w, w2pk,  8, 20,  96, 32);
    pack_w <<<(192 * 64 + 255) / 256, 256, 0, stream>>>(off1_w, wo1pk, 20, 54, 192, 64);
    pack_dw<<<(192 * 48 + 255) / 256, 256, 0, stream>>>(d1_w,   wd1pk, 20, 2, 40, 192, 48);
    pack_w <<<(384 * 64 + 255) / 256, 256, 0, stream>>>(off2_w, wo2pk, 40, 54, 384, 64);
    pack_dw<<<(384 * 64 + 255) / 256, 256, 0, stream>>>(d2_w,   wd2pk, 40, 2, 64, 384, 64);

    // ---- stage 1: conv1 + relu -----------------------------------------
    nchw_to_nhwc3<<<(NPIX * 3 + 255) / 256, 256, 0, stream>>>(x, xh);
    conv_gemm<3, 8, true, false><<<NPIX / 64, 256, 0, stream>>>(xh, nullptr, w1pk, nullptr, a1);
    hipMemsetAsync(sums, 0, 2 * 64 * 4, stream);
    stats_accum<8><<<1024, 256, 0, stream>>>(a1, sums);
    stats_final<8><<<1, 64, 0, stream>>>(sums, bn1_g, bn1_b, bnp1);

    // ---- stage 2: conv2 + bias + relu (bn1 fused on input) -------------
    conv_gemm<8, 20, true, true><<<NPIX / 64, 256, 0, stream>>>(a1, bnp1, w2pk, conv2_b, a2);
    hipMemsetAsync(sums, 0, 2 * 64 * 4, stream);
    stats_accum<20><<<1024, 256, 0, stream>>>(a2, sums);
    stats_final<20><<<1, 64, 0, stream>>>(sums, bn2_g, bn2_b, bnp2);

    // ---- stage 3: DCNv2 #1 (offset conv + deformable GEMM) -------------
    conv_gemm<20, 54, false, true><<<NPIX / 64, 256, 0, stream>>>(a2, bnp2, wo1pk, off1_b, om1);
    dconv_gemm<20, 2, 40><<<NPIX / 64, 256, 0, stream>>>(a2, bnp2, om1, wd1pk, d1_b, a3);
    hipMemsetAsync(sums, 0, 2 * 64 * 4, stream);
    stats_accum<40><<<1024, 256, 0, stream>>>(a3, sums);
    stats_final<40><<<1, 64, 0, stream>>>(sums, bn3_g, bn3_b, bnp3);

    // ---- stage 4: DCNv2 #2 ---------------------------------------------
    conv_gemm<40, 54, false, true><<<NPIX / 64, 256, 0, stream>>>(a3, bnp3, wo2pk, off2_b, om2);
    dconv_gemm<40, 2, 64><<<NPIX / 64, 256, 0, stream>>>(a3, bnp3, om2, wd2pk, d2_b, a4);
    hipMemsetAsync(sums, 0, 2 * 64 * 4, stream);
    stats_accum<64><<<1024, 256, 0, stream>>>(a4, sums);
    stats_final<64><<<1, 64, 0, stream>>>(sums, bn4_g, bn4_b, bnp4);

    // ---- epilogue: em (NCHW, bn4 applied) + pool/fc/softmax ------------
    emit_em<<<(NPIX * 64) / 256, 256, 0, stream>>>(a4, bnp4, out_em);
    pool_fc<<<BATCH, 256, 0, stream>>>(out_em, fc_w, fc_b, out_sm);
}
